// UncertaintyEstimatorCLS_34600256537501
// MI455X (gfx1250) — compile-verified, matrix operand order checked by reading, not run on hardware
//
#include <hip/hip_runtime.h>
#include <hip/hip_bf16.h>
#include <math.h>

// ---------------------------------------------------------------------------
// UncertaintyEstimatorCLS fused for MI455X (gfx1250, wave32).
// Kernel 1: per-box entropy via v_wmma_f32_16x16x32_f16 ones-reduction,
//           branchless b128 operand loads (clamped+masked), raw v_log_f32,
//           vectorized b128 result stores.
// Kernel 1b: scalar fallback for C not multiple of 8 (never hit for C=80).
// Kernel 2: fused first-match IoU scan (ballot+ctz) + K-average.
// ---------------------------------------------------------------------------

typedef __attribute__((ext_vector_type(16))) _Float16 v16h;
typedef __attribute__((ext_vector_type(8)))  float    v8f;

static __device__ __forceinline__ unsigned int ballot32(bool p) {
#if __has_builtin(__builtin_amdgcn_ballot_w32)
  return __builtin_amdgcn_ballot_w32(p);
#else
  return (unsigned int)__ballot((int)p);
#endif
}

// -p * log2(p): raw v_log_f32, no denormal fixup (softmax => p normal, >0)
static __device__ __forceinline__ float neg_plog2p(float p) {
  return -p * __builtin_amdgcn_logf(p);
}

// --------------------------------------------------------------------------
// Kernel 1: u_all[b] = 1 - H(conf_row_b)/ln(C) for all K*M boxes.
// One wave per 16-box tile. A (16x32 f16) holds t = -p*log2(p) chunks of the
// class axis; B = all-ones, so D[m][n] = row-sum = H/ln2. Accumulate 3 chunks
// (K=96 >= C=80, zero padded), then u = 1 - acc * (ln2/lnC).
// Requires C % 8 == 0 and 8 <= C <= 96 (host dispatches scalar otherwise).
// ISA A layout (16-bit, 16x32): lane L<16 -> M=L, elems 0..7 = K 0..7,
// elems 8..15 = K 16..23; lane L>=16 -> M=L-16, K 8..15 and 24..31.
// --------------------------------------------------------------------------
__global__ void entropy_wmma_kernel(const float* __restrict__ confs,
                                    float* __restrict__ u_all,
                                    int total_boxes, int C, float inv_log2C) {
  const int lane  = threadIdx.x & 31;
  const int tile  = blockIdx.x * (blockDim.x >> 5) + (threadIdx.x >> 5);
  const int ntile = (total_boxes + 15) >> 4;
  if (tile >= ntile) return;                 // wave-uniform: EXEC stays all-1s

  const int b0    = tile << 4;
  const int m     = lane & 15;               // matrix row this lane feeds
  const int khalf = (lane < 16) ? 0 : 8;     // which K-halves this lane holds
  int box = b0 + m;
  if (box >= total_boxes) box = total_boxes - 1;  // clamp: EXEC stays all-1s
  const float* __restrict__ row = confs + (size_t)box * (size_t)C;

  v16h ones;
#pragma unroll
  for (int e = 0; e < 16; ++e) ones[e] = (_Float16)1.0f;

  v8f acc = {};
#pragma unroll
  for (int chunk = 0; chunk < 3; ++chunk) {  // K = 3*32 = 96 >= C, zero pad
    const int kb = chunk * 32;
    v16h a;
#pragma unroll
    for (int half = 0; half < 2; ++half) {   // elems 0..7 then 8..15
      const int start = kb + khalf + half * 16;
      const bool valid = (start + 8 <= C);
      // Clamp the load address into bounds (C-8 is 8-aligned since C%8==0);
      // loads always execute -> no exec-mask branches, values masked below.
      const int s = valid ? start : (C - 8);
      const float4* __restrict__ r4 = (const float4*)(row + s);
      const float4 p0 = r4[0];
      const float4 p1 = r4[1];
      const float pv[8] = {p0.x, p0.y, p0.z, p0.w, p1.x, p1.y, p1.z, p1.w};
#pragma unroll
      for (int j = 0; j < 8; ++j)
        a[half * 8 + j] = valid ? (_Float16)neg_plog2p(pv[j]) : (_Float16)0.0f;
    }
    acc = __builtin_amdgcn_wmma_f32_16x16x32_f16(
        false, a, false, ones, (short)0, acc, false, false);
  }

  // D layout: VGPR v, lanes 0-15 -> (M=v, N=lane); lanes 16-31 -> (M=8+v).
  // Lane 0 (N=0) has sums for M=0..7; lane 16 for M=8..15. Each writes 8
  // consecutive floats; b0+mbase is 8-aligned -> two b128 stores (full tile).
  if ((lane & 15) == 0) {
    const int mbase = (lane < 16) ? 0 : 8;
    float r[8];
#pragma unroll
    for (int v = 0; v < 8; ++v) r[v] = 1.0f - acc[v] * inv_log2C;

    if (b0 + 16 <= total_boxes) {            // full tile: vector stores
      float4* __restrict__ o4 = (float4*)(u_all + b0 + mbase);
      o4[0] = make_float4(r[0], r[1], r[2], r[3]);
      o4[1] = make_float4(r[4], r[5], r[6], r[7]);
    } else {                                 // partial tail tile
#pragma unroll
      for (int v = 0; v < 8; ++v) {
        const int b = b0 + mbase + v;
        if (b < total_boxes) u_all[b] = r[v];
      }
    }
  }
}

// --------------------------------------------------------------------------
// Kernel 1b: generic-C scalar fallback (one thread per box, tight loop).
// --------------------------------------------------------------------------
__global__ void entropy_scalar_kernel(const float* __restrict__ confs,
                                      float* __restrict__ u_all,
                                      int total_boxes, int C, float inv_log2C) {
  const int b = blockIdx.x * blockDim.x + threadIdx.x;
  if (b >= total_boxes) return;
  const float* __restrict__ row = confs + (size_t)b * (size_t)C;
  float h = 0.0f;                            // sum p*log2(p) = -H/ln2
  for (int k = 0; k < C; ++k) {
    const float p = row[k];
    h += p * __builtin_amdgcn_logf(p);
  }
  u_all[b] = 1.0f + h * inv_log2C;
}

// --------------------------------------------------------------------------
// Kernel 2: one wave per pred n. For each pass k, lanes scan 32 boxes/iter;
// ballot+ctz gives the FIRST IoU>0.5 match (exact break semantics).
// --------------------------------------------------------------------------
__global__ void match_accum_kernel(const float* __restrict__ pred,
                                   const float* __restrict__ dboxes,
                                   const float* __restrict__ u_all,
                                   float* __restrict__ out,
                                   int N, int M, int K) {
  const int lane = threadIdx.x & 31;
  const int n    = blockIdx.x * (blockDim.x >> 5) + (threadIdx.x >> 5);
  if (n >= N) return;

  // pred row stride 6 floats = 24B -> 8B aligned: two b64 loads
  const float2* __restrict__ p2 = (const float2*)(pred + (size_t)n * 6);
  const float2 plo = p2[0], phi = p2[1];
  const float px1 = plo.x, py1 = plo.y, px2 = phi.x, py2 = phi.y;
  const float parea = (px2 - px1) * (py2 - py1);

  float sum = 0.0f, cnt = 0.0f;
  for (int k = 0; k < K; ++k) {
    const float* __restrict__ base = dboxes + (size_t)k * (size_t)M * 6;
    int first = -1;
    for (int mb = 0; mb < M; mb += 32) {
      const int m = mb + lane;
      bool hit = false;
      if (m < M) {
        const float2* __restrict__ b2 = (const float2*)(base + (size_t)m * 6);
        if (mb + 32 < M)  // stage next chunk -> global_prefetch_b8
          __builtin_prefetch(base + (size_t)(m + 32) * 6, 0, 1);
        const float2 blo = b2[0], bhi = b2[1];
        const float bx1 = blo.x, by1 = blo.y, bx2 = bhi.x, by2 = bhi.y;
        const float ix1 = fmaxf(px1, bx1), iy1 = fmaxf(py1, by1);
        const float ix2 = fminf(px2, bx2), iy2 = fminf(py2, by2);
        const float iw = fmaxf(ix2 - ix1, 0.0f), ih = fmaxf(iy2 - iy1, 0.0f);
        const float inter = iw * ih;
        const float barea = (bx2 - bx1) * (by2 - by1);
        const float iou = inter / (parea + barea - inter + 1e-7f);
        hit = (iou > 0.5f);
      }
      const unsigned int bal = ballot32(hit);   // wave-uniform result
      if (bal != 0u) { first = mb + (int)__builtin_ctz(bal); break; }
    }
    if (first >= 0) { sum += u_all[k * M + first]; cnt += 1.0f; }
  }

  if (lane == 0)
    out[n] = (cnt > 0.0f) ? (sum / cnt) : __int_as_float(0x7fc00000);
}

// --------------------------------------------------------------------------
extern "C" void kernel_launch(void* const* d_in, const int* in_sizes, int n_in,
                              void* d_out, int out_size, void* d_ws, size_t ws_size,
                              hipStream_t stream) {
  const float* pred  = (const float*)d_in[0];   // [N,6]
  const float* dpred = (const float*)d_in[1];   // [K,M,6]
  const float* dconf = (const float*)d_in[2];   // [K,M,C]
  float* out = (float*)d_out;

  const int N           = out_size;                  // 1000
  const int total_boxes = in_sizes[1] / 6;           // K*M = 20000
  const int C           = in_sizes[2] / total_boxes; // 80
  const int K           = 10;                        // fixed by reference
  const int M           = total_boxes / K;           // 2000

  float* u_all = (float*)d_ws;                       // K*M floats (80 KB)
  // acc holds H/ln2; u = 1 - acc * (ln2/lnC) = 1 - acc / log2(C)
  const float inv_log2C = logf(2.0f) / logf((float)C);

  const int wavesPerBlock = 8;                       // 256 threads = 8 wave32
  if ((C % 8) == 0 && C >= 8 && C <= 96) {
    const int ntiles = (total_boxes + 15) / 16;
    const int gridA  = (ntiles + wavesPerBlock - 1) / wavesPerBlock;
    entropy_wmma_kernel<<<gridA, 256, 0, stream>>>(dconf, u_all, total_boxes,
                                                   C, inv_log2C);
  } else {
    const int gridA = (total_boxes + 255) / 256;
    entropy_scalar_kernel<<<gridA, 256, 0, stream>>>(dconf, u_all, total_boxes,
                                                     C, inv_log2C);
  }

  const int gridB = (N + wavesPerBlock - 1) / wavesPerBlock;
  match_accum_kernel<<<gridB, 256, 0, stream>>>(pred, dpred, u_all, out,
                                                N, M, K);
}
